// LSTMCapsNet_27788438405567
// MI455X (gfx1250) — matrix operand order repaired and checked by reading, not compiled, not run on hardware
//
#include <hip/hip_runtime.h>
#include <math.h>

typedef __attribute__((ext_vector_type(16))) _Float16 v16h;
typedef __attribute__((ext_vector_type(8)))  float    v8f;

#define B_    128
#define L_    100
#define IND   160     // word 100 + tag 20 + pos1 20 + pos2 20
#define HID_  256
#define G4    1024    // 4*HID
#define NCAP  16
#define CAPD  16
#define NCLS  19
#define ICAP  1600    // L_*NCAP
#define NOUT  304     // NCLS*CAPD
#define EMARK 68
#define FRAG  512     // halfs per packed 32x16 B fragment (32 lanes * 16)

// ---------------- WMMA fragment helpers (CDNA5 16x16x32 f16, f32 acc) -------

static __device__ __forceinline__ int laneId() { return (int)(threadIdx.x & 31); }

// A: 16x32 (MxK) f16, row-major source with stride lda.
// lanes 0-15: row M=lane, K 0-7 (v0-3) and 16-23 (v4-7)
// lanes 16-31: row M=lane-16, K 8-15 and 24-31
static __device__ __forceinline__ v16h load_a_f16(const _Float16* p, int lda) {
  int lane = laneId();
  int m  = lane & 15;
  int kb = (lane >> 4) << 3;                 // 0 or 8
  const _Float16* r = p + (size_t)m * lda;
  v16h a;
#pragma unroll
  for (int j = 0; j < 8; ++j) a[j] = r[kb + j];
#pragma unroll
  for (int j = 0; j < 8; ++j) a[8 + j] = r[16 + kb + j];
  return a;
}

// B fragment pre-packed in register layout: [lane][16 halfs] contiguous.
static __device__ __forceinline__ v16h load_b_packed(const _Float16* frag) {
  return ((const v16h*)frag)[laneId()];
}

static __device__ __forceinline__ v8f wmma_f16(v16h a, v16h b, v8f c) {
  return __builtin_amdgcn_wmma_f32_16x16x32_f16(false, a, false, b, (short)0, c,
                                                false, false);
}

// ---------------- kernels ----------------------------------------------------

// gather embeddings, concat, convert to f16 : embh [B*L][160]
__global__ void lc_embed(const int* word, const int* tag, const int* pos1,
                         const int* pos2, const float* wemb, const float* temb,
                         const float* p1emb, const float* p2emb, _Float16* embh) {
  int idx = blockIdx.x * blockDim.x + threadIdx.x;
  if (idx >= B_ * L_ * IND) return;
  int k   = idx % IND;
  int row = idx / IND;
  float v;
  if (k < 100)      v = wemb[(size_t)word[row] * 100 + k];
  else if (k < 120) v = temb[(size_t)tag[row] * 20 + (k - 100)];
  else if (k < 140) v = p1emb[(size_t)pos1[row] * 20 + (k - 120)];
  else              v = p2emb[(size_t)pos2[row] * 20 + (k - 140)];
  embh[idx] = (_Float16)v;
}

// Pack LSTM weight w [Ntot][Ktot] f32 (torch layout: gates x input) into
// B fragments: packed[((kt*ntTiles + nt))*512 + lane*16 + j] where the
// fragment covers K = kt*32 + (j>=8 ? 16 : 0) + (lane>=16 ? 8 : 0) + j%8, N = nt*16 + lane%16.
__global__ void lc_pack_b(const float* w, _Float16* packed, int Ktot,
                          int ktTiles, int ntTiles) {
  int idx = blockIdx.x * blockDim.x + threadIdx.x;
  int total = ktTiles * ntTiles * FRAG;
  if (idx >= total) return;
  int j    = idx & 15;
  int lane = (idx >> 4) & 31;
  int frag = idx >> 9;
  int kt = frag / ntTiles;
  int nt = frag % ntTiles;
  int n  = nt * 16 + (lane & 15);
  int K  = kt * 32 + ((j >> 3) << 4) + ((lane >> 4) << 3) + (j & 7);
  packed[idx] = (_Float16)w[(size_t)n * Ktot + K];
}

// W_caps [1600][16][304] f32 -> packed fragments [1600*19][512] f16,
// K rows 16..31 zero-padded.
__global__ void lc_caps_pack(const float* W, _Float16* packed) {
  int idx = blockIdx.x * blockDim.x + threadIdx.x;
  if (idx >= ICAP * NCLS * FRAG) return;
  int j    = idx & 15;
  int lane = (idx >> 4) & 31;
  int frag = idx >> 9;
  int i  = frag / NCLS;
  int nt = frag % NCLS;
  int n  = nt * 16 + (lane & 15);
  int K  = ((j >> 3) << 4) + ((lane >> 4) << 3) + (j & 7);
  packed[idx] = (K < CAPD) ? (_Float16)W[((size_t)i * CAPD + K) * NOUT + n]
                           : (_Float16)0.f;
}

// xg = embh[12800x160] * w_ih^T + (b_ih + b_hh)
// One wave computes a 16x64 strip (4 N-tiles), A fragment reused 4x per K-step.
__global__ void lc_gemm_xg(const _Float16* A, const _Float16* Bp,
                           const float* bih, const float* bhh, float* xg) {
  int tile = blockIdx.x * 4 + (threadIdx.x >> 5);   // 800*16 = 12800 strips
  int mt = tile >> 4;
  int ng = tile & 15;                               // N-group of 4 tiles
  v8f acc[4] = {};
#pragma unroll
  for (int kt = 0; kt < 5; ++kt) {                  // K = 160 = 5*32
    v16h a = load_a_f16(A + (size_t)(mt * 16) * IND + kt * 32, IND);
#pragma unroll
    for (int q = 0; q < 4; ++q) {
      v16h b = load_b_packed(Bp + (size_t)(kt * 64 + ng * 4 + q) * FRAG);
      acc[q] = wmma_f16(a, b, acc[q]);
    }
  }
  int lane = laneId();
  int m0 = mt * 16 + ((lane >> 4) << 3);
#pragma unroll
  for (int q = 0; q < 4; ++q) {
    int n = (ng * 4 + q) * 16 + (lane & 15);
    float bias = bih[n] + bhh[n];
#pragma unroll
    for (int r = 0; r < 8; ++r)
      xg[(size_t)(m0 + r) * G4 + n] = acc[q][r] + bias;
  }
}

// g = h16[128x256] * w_hh^T + xg[:, t, :]
// B fragments double-buffered into LDS with async copies (gfx1250 path).
__global__ void lc_hh_gemm(const _Float16* h16, const _Float16* Bp,
                           const float* xg, int t, float* g) {
  __shared__ __align__(32) _Float16 stage[4][2][FRAG];   // 4 waves x 2 x 1KB
  int w    = threadIdx.x >> 5;
  int lane = laneId();
  int tile = blockIdx.x * 4 + w;                    // 8*64 = 512 tiles
  int mt = tile >> 6;
  int nt = tile & 63;
  unsigned int lds0 =
      (unsigned int)(uintptr_t)(&stage[w][0][0]) + (unsigned int)(lane * 32);
  const char* bbase = (const char*)Bp;

  auto issue = [&](int kt, int buf) {
    unsigned long long ga =
        (unsigned long long)(uintptr_t)(bbase +
                                        (size_t)(kt * 64 + nt) * FRAG * 2) +
        (unsigned int)(lane * 32);
    unsigned int la = lds0 + (unsigned int)(buf * (FRAG * 2));
    asm volatile("global_load_async_to_lds_b128 %0, %1, off\n\t"
                 "global_load_async_to_lds_b128 %0, %1, off offset:16"
                 :: "v"(la), "v"(ga) : "memory");
  };

  issue(0, 0);
  v8f acc = {};
#pragma unroll
  for (int kt = 0; kt < 8; ++kt) {                  // K = 256 = 8*32
    if (kt < 7) {
      issue(kt + 1, (kt + 1) & 1);
      asm volatile("s_wait_asynccnt 0x2" ::: "memory");  // kt's pair done
    } else {
      asm volatile("s_wait_asynccnt 0x0" ::: "memory");
    }
    v16h a = load_a_f16(h16 + (size_t)(mt * 16) * HID_ + kt * 32, HID_);
    v16h b = *(const v16h*)(&stage[w][kt & 1][lane * 16]);
    acc = wmma_f16(a, b, acc);
  }
  int n  = nt * 16 + (lane & 15);
  int m0 = mt * 16 + ((lane >> 4) << 3);
#pragma unroll
  for (int r = 0; r < 8; ++r) {
    int bi = m0 + r;
    g[(size_t)bi * G4 + n] = acc[r] + xg[((size_t)bi * L_ + t) * G4 + n];
  }
}

// LSTM pointwise gates: torch order i,f,g,o
__global__ void lc_gate(const float* g, float* c, _Float16* h16, float* hout, int t) {
  int idx = blockIdx.x * blockDim.x + threadIdx.x;
  if (idx >= B_ * HID_) return;
  int b = idx / HID_, j = idx % HID_;
  const float* gr = g + (size_t)b * G4;
  float gi = gr[j], gf = gr[HID_ + j], gg = gr[2 * HID_ + j], go = gr[3 * HID_ + j];
  float si = 1.f / (1.f + expf(-gi));
  float sf = 1.f / (1.f + expf(-gf));
  float so = 1.f / (1.f + expf(-go));
  float cn = sf * c[idx] + si * tanhf(gg);
  float hn = so * tanhf(cn);
  c[idx]  = cn;
  h16[idx] = (_Float16)hn;
  hout[((size_t)b * L_ + t) * HID_ + j] = hn;
}

__global__ void lc_sum(const float* hf, const float* hb, float* x) {
  int idx = blockIdx.x * blockDim.x + threadIdx.x;
  if (idx >= B_ * L_ * HID_) return;
  x[idx] = hf[idx] + hb[idx];
}

// first index where pos==68 per row, then he = x[e1] + x[e2]
__global__ void lc_entity(const int* pos1, const int* pos2, const float* x, float* he) {
  __shared__ int e1s, e2s;
  int b = blockIdx.x, tid = threadIdx.x;          // 256 threads
  if (tid == 0) { e1s = L_ - 1; e2s = L_ - 1; }
  __syncthreads();
  if (tid < L_) {
    if (pos1[b * L_ + tid] == EMARK) atomicMin(&e1s, tid);
    if (pos2[b * L_ + tid] == EMARK) atomicMin(&e2s, tid);
  }
  __syncthreads();
  int e1 = e1s, e2 = e2s;
  he[(size_t)b * HID_ + tid] = x[((size_t)b * L_ + e1) * HID_ + tid] +
                               x[((size_t)b * L_ + e2) * HID_ + tid];
}

// att = softmax_l( x[b,l,:] . he[b,:] )
__global__ void lc_att(const float* x, const float* he, float* att) {
  __shared__ float lg[L_];
  int b = blockIdx.x, tid = threadIdx.x;          // 128 threads
  if (tid < L_) {
    const float* xr = x + ((size_t)b * L_ + tid) * HID_;
    const float* hr = he + (size_t)b * HID_;
    float s = 0.f;
    for (int d = 0; d < HID_; ++d) s += xr[d] * hr[d];
    lg[tid] = s;
  }
  __syncthreads();
  if (tid == 0) {
    float mx = lg[0];
    for (int l = 1; l < L_; ++l) mx = fmaxf(mx, lg[l]);
    float se = 0.f;
    for (int l = 0; l < L_; ++l) se += expf(lg[l] - mx);
    for (int l = 0; l < L_; ++l) att[b * L_ + l] = expf(lg[l] - mx) / se;
  }
}

// u = squash(x reshaped [B][1600][16])
__global__ void lc_squash_u(const float* x, float* u) {
  int idx = blockIdx.x * blockDim.x + threadIdx.x;
  if (idx >= B_ * ICAP) return;
  int b = idx / ICAP, i = idx % ICAP;
  int l = i >> 4, cap = i & 15;
  const float* src = x + ((size_t)b * L_ + l) * HID_ + cap * CAPD;
  float n2 = 0.f;
#pragma unroll
  for (int d = 0; d < CAPD; ++d) n2 += src[d] * src[d];
  float sc = n2 / (1.f + n2) / sqrtf(n2 + 1e-9f);
  float* dst = u + (size_t)idx * CAPD;
#pragma unroll
  for (int d = 0; d < CAPD; ++d) dst[d] = src[d] * sc;
}

// u_hat[b][i][304] = u[b][i][16] * W_caps[i]   (K padded 16->32, packed B)
// One wave owns (i, mt) and sweeps all 19 N-tiles, reusing the A fragment.
__global__ void lc_uhat(const float* u, const _Float16* Wp, float* uhat) {
  int tile = blockIdx.x * 4 + (threadIdx.x >> 5);  // 1600*8 = 12800 strips
  int i  = tile >> 3;
  int mt = tile & 7;
  int lane = laneId();
  int m  = lane & 15;
  int kb = (lane >> 4) << 3;
  const float* ar = u + ((size_t)(mt * 16 + m) * ICAP + i) * CAPD;
  v16h a;
#pragma unroll
  for (int j = 0; j < 8; ++j) a[j] = (_Float16)ar[kb + j];   // K 0..15 real
#pragma unroll
  for (int j = 0; j < 8; ++j) a[8 + j] = (_Float16)0.f;      // K 16..31 zero
  int n0 = lane & 15;
  int m0 = mt * 16 + ((lane >> 4) << 3);
#pragma unroll 1
  for (int nt = 0; nt < NCLS; ++nt) {
    v16h bf = load_b_packed(Wp + (size_t)(i * NCLS + nt) * FRAG);
    v8f acc = {};
    acc = wmma_f16(a, bf, acc);
    int n = nt * 16 + n0;
#pragma unroll
    for (int r = 0; r < 8; ++r)
      uhat[((size_t)(m0 + r) * ICAP + i) * NOUT + n] = acc[r];
  }
}

__global__ void lc_bb_init(const float* brt, float* bb) {
  int idx = blockIdx.x * blockDim.x + threadIdx.x;
  if (idx >= B_ * ICAP * NCLS) return;
  bb[idx] = brt[idx % (ICAP * NCLS)];
}

// c = softmax_o(bb)*alpha ; s = sum_i c*u_hat ; v = squash(s)
__global__ void lc_route_sc(const float* bb, const float* att, const float* uhat,
                            float* v) {
  __shared__ float acc[CAPD];
  int b = blockIdx.x / NCLS;
  int o = blockIdx.x % NCLS;
  int tid = threadIdx.x;                           // 128 threads
  if (tid < CAPD) acc[tid] = 0.f;
  __syncthreads();
  float loc[CAPD];
#pragma unroll
  for (int d = 0; d < CAPD; ++d) loc[d] = 0.f;
  for (int i = tid; i < ICAP; i += 128) {
    const float* br = bb + ((size_t)b * ICAP + i) * NCLS;
    float mx = br[0];
    for (int q = 1; q < NCLS; ++q) mx = fmaxf(mx, br[q]);
    float se = 0.f;
    for (int q = 0; q < NCLS; ++q) se += expf(br[q] - mx);
    float cio = expf(br[o] - mx) / se * att[b * L_ + (i >> 4)];
    const float* uh = uhat + ((size_t)b * ICAP + i) * NOUT + o * CAPD;
#pragma unroll
    for (int d = 0; d < CAPD; ++d) loc[d] += cio * uh[d];
  }
#pragma unroll
  for (int d = 0; d < CAPD; ++d) atomicAdd(&acc[d], loc[d]);
  __syncthreads();
  if (tid == 0) {
    float n2 = 0.f;
    for (int d = 0; d < CAPD; ++d) n2 += acc[d] * acc[d];
    float sc = n2 / (1.f + n2) / sqrtf(n2 + 1e-9f);
    for (int d = 0; d < CAPD; ++d)
      v[((size_t)b * NCLS + o) * CAPD + d] = acc[d] * sc;
  }
}

// bb += u_hat . v
__global__ void lc_route_bb(float* bb, const float* uhat, const float* v) {
  int idx = blockIdx.x * blockDim.x + threadIdx.x;
  if (idx >= B_ * ICAP * NCLS) return;
  int o  = idx % NCLS;
  int t2 = idx / NCLS;
  int i  = t2 % ICAP;
  int b  = t2 / ICAP;
  const float* uh = uhat + ((size_t)b * ICAP + i) * NOUT + o * CAPD;
  const float* vr = v + ((size_t)b * NCLS + o) * CAPD;
  float s = 0.f;
#pragma unroll
  for (int d = 0; d < CAPD; ++d) s += uh[d] * vr[d];
  bb[idx] += s;
}

__global__ void lc_out(const float* v, float* out) {
  int idx = blockIdx.x * blockDim.x + threadIdx.x;
  if (idx >= B_ * NCLS) return;
  const float* vr = v + (size_t)idx * CAPD;
  float n2 = 0.f;
#pragma unroll
  for (int d = 0; d < CAPD; ++d) n2 += vr[d] * vr[d];
  out[idx] = sqrtf(n2 + 1e-9f);
}

// ---------------- host orchestration ----------------------------------------

extern "C" void kernel_launch(void* const* d_in, const int* in_sizes, int n_in,
                              void* d_out, int out_size, void* d_ws, size_t ws_size,
                              hipStream_t stream) {
  const int*   word  = (const int*)d_in[0];
  const int*   tag   = (const int*)d_in[1];
  const int*   pos1  = (const int*)d_in[2];
  const int*   pos2  = (const int*)d_in[3];
  const float* wemb  = (const float*)d_in[4];
  const float* temb  = (const float*)d_in[5];
  const float* p1emb = (const float*)d_in[6];
  const float* p2emb = (const float*)d_in[7];
  const float* wihf  = (const float*)d_in[8];
  const float* whhf  = (const float*)d_in[9];
  const float* bihf  = (const float*)d_in[10];
  const float* bhhf  = (const float*)d_in[11];
  const float* wihb  = (const float*)d_in[12];
  const float* whhb  = (const float*)d_in[13];
  const float* bihb  = (const float*)d_in[14];
  const float* bhhb  = (const float*)d_in[15];
  const float* Wcaps = (const float*)d_in[16];
  const float* brt   = (const float*)d_in[17];
  float* out = (float*)d_out;

  char* cur = (char*)d_ws;
  auto alloc = [&](size_t bytes) -> void* {
    void* p = (void*)cur;
    cur += (bytes + 255) & ~(size_t)255;
    return p;
  };
  _Float16* embh  = (_Float16*)alloc((size_t)B_ * L_ * IND * 2);
  _Float16* wihPF = (_Float16*)alloc((size_t)5 * 64 * FRAG * 2);
  _Float16* wihPB = (_Float16*)alloc((size_t)5 * 64 * FRAG * 2);
  _Float16* whhPF = (_Float16*)alloc((size_t)8 * 64 * FRAG * 2);
  _Float16* whhPB = (_Float16*)alloc((size_t)8 * 64 * FRAG * 2);
  float*    xgF   = (float*)alloc((size_t)B_ * L_ * G4 * 4);
  float*    xgB   = (float*)alloc((size_t)B_ * L_ * G4 * 4);
  _Float16* h16F  = (_Float16*)alloc((size_t)B_ * HID_ * 2);
  _Float16* h16B  = (_Float16*)alloc((size_t)B_ * HID_ * 2);
  float*    cF    = (float*)alloc((size_t)B_ * HID_ * 4);
  float*    cB    = (float*)alloc((size_t)B_ * HID_ * 4);
  float*    gF    = (float*)alloc((size_t)B_ * G4 * 4);
  float*    gB    = (float*)alloc((size_t)B_ * G4 * 4);
  float*    hf    = (float*)alloc((size_t)B_ * L_ * HID_ * 4);
  float*    hb    = (float*)alloc((size_t)B_ * L_ * HID_ * 4);
  float*    x     = (float*)alloc((size_t)B_ * L_ * HID_ * 4);
  float*    he    = (float*)alloc((size_t)B_ * HID_ * 4);
  float*    att   = (float*)alloc((size_t)B_ * L_ * 4);
  float*    u     = (float*)alloc((size_t)B_ * ICAP * CAPD * 4);
  _Float16* WcP   = (_Float16*)alloc((size_t)ICAP * NCLS * FRAG * 2);
  float*    uhat  = (float*)alloc((size_t)B_ * ICAP * NOUT * 4);
  float*    bb    = (float*)alloc((size_t)B_ * ICAP * NCLS * 4);
  float*    vbuf  = (float*)alloc((size_t)B_ * NCLS * CAPD * 4);

  // 1) embeddings -> f16
  lc_embed<<<(B_ * L_ * IND + 255) / 256, 256, 0, stream>>>(
      word, tag, pos1, pos2, wemb, temb, p1emb, p2emb, embh);

  // 2) weight prep: pack into WMMA B-fragment layout (f16)
  lc_pack_b<<<(5 * 64 * FRAG + 255) / 256, 256, 0, stream>>>(wihf, wihPF, IND, 5, 64);
  lc_pack_b<<<(5 * 64 * FRAG + 255) / 256, 256, 0, stream>>>(wihb, wihPB, IND, 5, 64);
  lc_pack_b<<<(8 * 64 * FRAG + 255) / 256, 256, 0, stream>>>(whhf, whhPF, HID_, 8, 64);
  lc_pack_b<<<(8 * 64 * FRAG + 255) / 256, 256, 0, stream>>>(whhb, whhPB, HID_, 8, 64);
  lc_caps_pack<<<(ICAP * NCLS * FRAG + 255) / 256, 256, 0, stream>>>(Wcaps, WcP);

  // 3) input-gate GEMMs (WMMA, 16x64 strip per wave)
  lc_gemm_xg<<<(800 * 16) / 4, 128, 0, stream>>>(embh, wihPF, bihf, bhhf, xgF);
  lc_gemm_xg<<<(800 * 16) / 4, 128, 0, stream>>>(embh, wihPB, bihb, bhhb, xgB);

  // 4) LSTM recurrence (both directions interleaved, h in f16 for WMMA)
  hipMemsetAsync(h16F, 0, (size_t)B_ * HID_ * 2, stream);
  hipMemsetAsync(h16B, 0, (size_t)B_ * HID_ * 2, stream);
  hipMemsetAsync(cF, 0, (size_t)B_ * HID_ * 4, stream);
  hipMemsetAsync(cB, 0, (size_t)B_ * HID_ * 4, stream);
  for (int t = 0; t < L_; ++t) {
    int tr = L_ - 1 - t;
    lc_hh_gemm<<<512 / 4, 128, 0, stream>>>(h16F, whhPF, xgF, t, gF);
    lc_gate<<<(B_ * HID_ + 255) / 256, 256, 0, stream>>>(gF, cF, h16F, hf, t);
    lc_hh_gemm<<<512 / 4, 128, 0, stream>>>(h16B, whhPB, xgB, tr, gB);
    lc_gate<<<(B_ * HID_ + 255) / 256, 256, 0, stream>>>(gB, cB, h16B, hb, tr);
  }

  // 5) x = hf + hb ; entity features ; attention ; primary caps
  lc_sum<<<(B_ * L_ * HID_ + 255) / 256, 256, 0, stream>>>(hf, hb, x);
  lc_entity<<<B_, 256, 0, stream>>>(pos1, pos2, x, he);
  lc_att<<<B_, 128, 0, stream>>>(x, he, att);
  lc_squash_u<<<(B_ * ICAP + 255) / 256, 256, 0, stream>>>(x, u);

  // 6) digit-caps predictions (WMMA, K padded 16->32, A reused over 19 N-tiles)
  lc_uhat<<<(ICAP * 8) / 4, 128, 0, stream>>>(u, WcP, uhat);

  // 7) routing
  lc_bb_init<<<(B_ * ICAP * NCLS + 255) / 256, 256, 0, stream>>>(brt, bb);
  for (int it = 0; it < 3; ++it) {
    lc_route_sc<<<B_ * NCLS, 128, 0, stream>>>(bb, att, uhat, vbuf);
    if (it < 2)
      lc_route_bb<<<(B_ * ICAP * NCLS + 255) / 256, 256, 0, stream>>>(bb, uhat, vbuf);
  }

  // 8) class probabilities = capsule lengths
  lc_out<<<(B_ * NCLS + 255) / 256, 256, 0, stream>>>(vbuf, out);
}
